// ProgressiveBJointBlock_37185826849134
// MI455X (gfx1250) — matrix operand order repaired and checked by reading, not compiled, not run on hardware
//
#include <hip/hip_runtime.h>
#include <hip/hip_bf16.h>
#include <math.h>

typedef __attribute__((ext_vector_type(16))) __bf16 v16bf;
typedef __attribute__((ext_vector_type(8)))  float  v8f;
typedef __attribute__((ext_vector_type(4)))  unsigned int v4u;
typedef __attribute__((ext_vector_type(8)))  int v8i;
typedef __attribute__((ext_vector_type(4)))  int v4i;

constexpr int   kD  = 256;
constexpr int   kB  = 4;
constexpr int   kS  = 8192;
constexpr int   kNE = 2048;
constexpr int   kNC = 512;
constexpr float kInvSqrtD = 0.0625f;   // 1/sqrt(256)
constexpr float kEps = 1e-5f;
constexpr float kScB  = 0.2f;          // ALPHA*B_SC
constexpr float kScX  = 0.075f;        // ALPHA*BB2S*X_SC
constexpr float kScSB = 0.075f;        // ALPHA*BS2B*X_SC

// ---------------- reductions (wave32) ----------------
__device__ __forceinline__ float waveSum(float v) {
#pragma unroll
  for (int o = 16; o > 0; o >>= 1) v += __shfl_xor(v, o, 32);
  return v;
}

__device__ __forceinline__ float blockSum256(float v, float* sm) {
  v = waveSum(v);
  const int lane = threadIdx.x & 31, w = threadIdx.x >> 5;
  if (lane == 0) sm[w] = v;
  __syncthreads();
  if (threadIdx.x == 0) {
    float s = 0.f;
#pragma unroll
    for (int i = 0; i < 8; ++i) s += sm[i];
    sm[0] = s;
  }
  __syncthreads();
  float r = sm[0];
  __syncthreads();
  return r;
}

__device__ __forceinline__ void atomAddF(float* p, float v) {
  __hip_atomic_fetch_add(p, v, __ATOMIC_RELAXED, __HIP_MEMORY_SCOPE_AGENT);
}

// ---------------- Tensor Data Mover: 2D tile -> LDS (D# per CDNA5 ISA) --
// tensor dims == tile dims (tile always fully in-bounds), data_size = 2B,
// pad_enable adds +4 DWORDs (8 halves) of LDS padding per tile row to keep the
// per-row LDS stride bank-friendly and 16B-aligned for ds_load_b128 gathers.
__device__ __forceinline__ void tdm_load_2d(const void* gptr, unsigned lds_off,
                                            unsigned tile0, unsigned tile1,
                                            unsigned long stride0,
                                            unsigned padIv, unsigned padAmt)
{
  unsigned long ga = (unsigned long)gptr;
  v4u g0;
  g0.x = 1u;                                           // count=1, user descriptor
  g0.y = lds_off;                                      // LDS byte address
  g0.z = (unsigned)(ga & 0xffffffffu);                 // global_addr[31:0]
  g0.w = (unsigned)((ga >> 32) & 0x01ffffffu) | (2u << 30);  // addr hi | type=2
  v8i g1;
  g1[0] = (int)((1u << 16) | (1u << 20) | (padIv << 22) | (padAmt << 25));
  g1[1] = (int)((tile0 & 0xffffu) << 16);              // tensor_dim0[15:0]
  g1[2] = (int)(((tile0 >> 16) & 0xffffu) | ((tile1 & 0xffffu) << 16));
  g1[3] = (int)(((tile1 >> 16) & 0xffffu) | ((tile0 & 0xffffu) << 16)); // tile_dim0
  g1[4] = (int)(tile1 & 0xffffu);                      // tile_dim1 (tile_dim2=0)
  g1[5] = (int)(unsigned)(stride0 & 0xffffffffu);      // tensor_dim0_stride
  g1[6] = (int)(unsigned)((stride0 >> 32) & 0xffffu);
  g1[7] = 0;
  v4i z4 = {0, 0, 0, 0};
#if __clang_major__ >= 23
  v8i z8 = {0, 0, 0, 0, 0, 0, 0, 0};
  __builtin_amdgcn_tensor_load_to_lds(g0, g1, z4, z4, z8, 0);
#else
  __builtin_amdgcn_tensor_load_to_lds(g0, g1, z4, z4, 0);
#endif
}

__device__ __forceinline__ unsigned ldsOffset(const void* p) {
  return (unsigned)(unsigned long)p;   // LDS aperture: low 32 bits = LDS offset
}

// ---------------- window propagation + tanh/LN update (bandwidth bound) ----
__global__ __launch_bounds__(256)
void k_window(const float* __restrict__ st, const float* __restrict__ vl,
              const float* __restrict__ w,  const float* __restrict__ g,
              const float* __restrict__ bb, float* __restrict__ ost,
              float* __restrict__ ovl)
{
  const int lane = threadIdx.x & 31;
  const int wid  = threadIdx.x >> 5;
  const long row = (long)blockIdx.x * 8 + wid;    // b*kS + n
  const int b = (int)(row / kS);
  const int n = (int)(row % kS);
  const float* bs = st + (long)b * kS * kD;
  const float* bv = vl + (long)b * kS * kD;
  const int d0 = lane * 8;

  float cs[8], q[8];
#pragma unroll
  for (int i = 0; i < 8; ++i) { cs[i] = bs[(long)n*kD + d0 + i]; q[i] = cs[i] * w[d0 + i]; }

  float sc[9];
#pragma unroll
  for (int k = 0; k < 9; ++k) {
    int m = n - 4 + k;
    bool valid = (m >= 0) && (m < kS);
    int mc = m < 0 ? 0 : (m >= kS ? kS - 1 : m);
    const float* r = bs + (long)mc*kD + d0;
    float p = 0.f;
#pragma unroll
    for (int i = 0; i < 8; ++i) p += q[i] * r[i];
    p = waveSum(p) * kInvSqrtD;
    sc[k] = valid ? p : -1e30f;
  }
  float mx = sc[0];
#pragma unroll
  for (int k = 1; k < 9; ++k) mx = fmaxf(mx, sc[k]);
  float a[9], ssum = 0.f;
#pragma unroll
  for (int k = 0; k < 9; ++k) { a[k] = expf(sc[k] - mx); ssum += a[k]; }
  float inv = 1.f / ssum;
#pragma unroll
  for (int k = 0; k < 9; ++k) a[k] *= inv;

  float ds[8] = {0,0,0,0,0,0,0,0}, dv[8] = {0,0,0,0,0,0,0,0};
#pragma unroll
  for (int k = 0; k < 9; ++k) {
    int m = n - 4 + k;
    int mc = m < 0 ? 0 : (m >= kS ? kS - 1 : m);
    const float* rs = bs + (long)mc*kD + d0;
    const float* rv = bv + (long)mc*kD + d0;
#pragma unroll
    for (int i = 0; i < 8; ++i) { ds[i] += a[k]*rs[i]; dv[i] += a[k]*rv[i]; }
  }
#pragma unroll
  for (int i = 0; i < 8; ++i) ost[row*kD + d0 + i] = tanhf(cs[i] + 0.25f*ds[i]);

  float y[8]; float s1 = 0.f;
#pragma unroll
  for (int i = 0; i < 8; ++i) { y[i] = bv[(long)n*kD + d0 + i] + 0.25f*dv[i]; s1 += y[i]; }
  float mean = waveSum(s1) * (1.f / kD);
  float s2 = 0.f;
#pragma unroll
  for (int i = 0; i < 8; ++i) { float c = y[i] - mean; s2 += c*c; }
  float var = waveSum(s2) * (1.f / kD);
  float rr = rsqrtf(var + kEps);
#pragma unroll
  for (int i = 0; i < 8; ++i) ovl[row*kD + d0 + i] = (y[i]-mean)*rr*g[d0+i] + bb[d0+i];
}

// ---------------- fused bf16-WMMA GEMM + running top-4 + softmax ----------
// A:  [B][N][256] bf16 rows (TDM-staged to LDS once per block).
// Bm: [M][256] bf16 rows; per-batch (bShared=0) or shared/pre-transposed W
// (bShared=1). B chunks are double-buffered via the Tensor Data Mover so the
// DMA of chunk k+1 overlaps the 4 WMMAs of chunk k. Logits never touch HBM:
// only per-row top-4 (idx, softmax weight) is emitted.
__global__ __launch_bounds__(128)
void k_gemm_topk(const __bf16* __restrict__ A, const __bf16* __restrict__ Bm,
                 const float* __restrict__ bias, float scale,
                 int N, int M, int bShared,
                 int* __restrict__ oIdx, float* __restrict__ oW)
{
  __shared__ __bf16 As[64][264];       // 256+8 halves: 528B stride, 16B aligned
  __shared__ __bf16 Bs[2][64][40];     // 32+8 halves: 80B stride, double buffer
  __shared__ float  Sc[64][65];

  const int b   = blockIdx.y;
  const int n0  = blockIdx.x * 64;
  const int tid = threadIdx.x;
  const int lane = tid & 31, wid = tid >> 5;

  const __bf16* Ab    = A + ((long)b * N + n0) * kD;
  const __bf16* Bbase = Bm + (bShared ? 0l : (long)b * M * kD);

  // stage the A tile (64 x 256 halves) with the TDM; row pad 8 halves
  if (wid == 0)
    tdm_load_2d(Ab, ldsOffset(&As[0][0]), 256u, 64u, 256ul, 6u, 3u);

  float b0 = -3.0e38f, b1 = -3.0e38f, b2 = -3.0e38f, b3 = -3.0e38f;
  int   i0 = 0, i1 = 0, i2 = 0, i3 = 0;

  for (int m0 = 0; m0 < M; m0 += 64) {
    v8f acc[4];
#pragma unroll
    for (int c = 0; c < 4; ++c)
#pragma unroll
      for (int r = 0; r < 8; ++r) acc[c][r] = 0.f;

    // kick off first B chunk of this M-tile (64 m-rows x 32 k-halves)
    if (wid == 0)
      tdm_load_2d(Bbase + (long)m0 * kD, ldsOffset(&Bs[0][0][0]),
                  32u, 64u, 256ul, 3u, 3u);

    for (int kc8 = 0; kc8 < 8; ++kc8) {
      const int kc = kc8 * 32;
      __syncthreads();   // everyone done reading the buffer TDM will overwrite
      if (wid == 0) {
        if (kc8 < 7)
          tdm_load_2d(Bbase + (long)m0 * kD + (kc + 32),
                      ldsOffset(&Bs[(kc8 + 1) & 1][0][0]),
                      32u, 64u, 256ul, 3u, 3u);
        if (kc8 < 7) __builtin_amdgcn_s_wait_tensorcnt(1);
        else         __builtin_amdgcn_s_wait_tensorcnt(0);
      }
      __syncthreads();   // chunk kc (and the A tile) visible to all waves

      // A fragment: row = wid*16 + (lane&15); K = {sel..sel+7, sel+16..sel+23}
      union { v16bf v; __bf16 e[16]; } af;
      const int sel  = (lane >> 4) * 8;
      const int arow = wid * 16 + (lane & 15);
#pragma unroll
      for (int j = 0; j < 8; ++j) {
        af.e[j]     = As[arow][kc + sel + j];
        af.e[8 + j] = As[arow][kc + sel + 16 + j];
      }
      // B fragments: col = c*16 + (lane&15); K = {kb..kb+7, kb+16..kb+23}
      const int kb  = (lane >> 4) * 8;
      const int buf = kc8 & 1;
#pragma unroll
      for (int c = 0; c < 4; ++c) {
        union { v16bf v; __bf16 e[16]; } bfm;
        const int col = c * 16 + (lane & 15);
#pragma unroll
        for (int j = 0; j < 16; ++j) bfm.e[j] = Bs[buf][col][kb + (j & 7) + ((j >> 3) << 4)];
        acc[c] = __builtin_amdgcn_wmma_f32_16x16x32_bf16(
            false, af.v, false, bfm.v, (short)0, acc[c], false, false);
      }
    }
    __syncthreads();

    // spill scores to LDS (scale + bias applied here)
    const int rbase = wid * 16 + (lane >> 4) * 8;
    const int col   = lane & 15;
#pragma unroll
    for (int c = 0; c < 4; ++c) {
      float bia = bias ? bias[m0 + c * 16 + col] : 0.f;
#pragma unroll
      for (int r = 0; r < 8; ++r)
        Sc[rbase + r][c * 16 + col] = acc[c][r] * scale + bia;
    }
    __syncthreads();

    // running top-4; strict '>' keeps lowest index on ties (lax.top_k semantics)
    if (tid < 64) {
      for (int j = 0; j < 64; ++j) {
        float v = Sc[tid][j];
        int   mi = m0 + j;
        if (v > b0)      { b3=b2;i3=i2; b2=b1;i2=i1; b1=b0;i1=i0; b0=v;i0=mi; }
        else if (v > b1) { b3=b2;i3=i2; b2=b1;i2=i1; b1=v;i1=mi; }
        else if (v > b2) { b3=b2;i3=i2; b2=v;i2=mi; }
        else if (v > b3) { b3=v;i3=mi; }
      }
    }
    __syncthreads();
  }

  if (tid < 64) {
    float e0 = 1.f, e1 = expf(b1 - b0), e2 = expf(b2 - b0), e3 = expf(b3 - b0);
    float inv = 1.f / (e0 + e1 + e2 + e3);
    long o = ((long)b * N + n0 + tid) * 4;
    oIdx[o+0]=i0; oIdx[o+1]=i1; oIdx[o+2]=i2; oIdx[o+3]=i3;
    oW[o+0]=e0*inv; oW[o+1]=e1*inv; oW[o+2]=e2*inv; oW[o+3]=e3*inv;
  }
}

// ---------------- scatter (sparse transitions) ----------------------------
__global__ __launch_bounds__(256)
void k_scatter(const float* __restrict__ src_s, const float* __restrict__ src_v,
               const int* __restrict__ idx, const float* __restrict__ wgt,
               float scale, int Nsrc, int Ndst,
               float* __restrict__ dst_s, float* __restrict__ dst_v)
{
  const long row = blockIdx.x;            // b*Nsrc + n
  const int  b   = (int)(row / Nsrc);
  const int  d   = threadIdx.x;
  const float s = src_s[row * kD + d];
  const float v = src_v[row * kD + d];
  const long o = row * 4;
#pragma unroll
  for (int j = 0; j < 4; ++j) {
    const int   m  = idx[o + j];
    const float ww = wgt[o + j] * scale;
    float* ps = dst_s + ((long)b * Ndst + m) * kD + d;
    float* pv = dst_v + ((long)b * Ndst + m) * kD + d;
    atomAddF(ps, ww * s);
    atomAddF(pv, ww * v);
  }
}

// ---------------- gather + tanh / LN epilogue (top-k propagation) ----------
__global__ __launch_bounds__(256)
void k_gather_combine(const float* __restrict__ old_s, const float* __restrict__ old_v,
                      const float* __restrict__ src_s, const float* __restrict__ src_v,
                      const int* __restrict__ idx, const float* __restrict__ wgt,
                      float sc, const float* __restrict__ g, const float* __restrict__ bb,
                      int N, float* __restrict__ out_s, float* __restrict__ out_v)
{
  __shared__ float sm[8];
  const long row = blockIdx.x;
  const int  b   = (int)(row / N);
  const int  d   = threadIdx.x;
  const long o   = row * 4;
  float ds = 0.f, dv = 0.f;
#pragma unroll
  for (int j = 0; j < 4; ++j) {
    const int   m = idx[o + j];
    const float w = wgt[o + j];
    const long  base = ((long)b * N + m) * kD + d;
    ds += w * src_s[base];
    dv += w * src_v[base];
  }
  out_s[row * kD + d] = tanhf(old_s[row * kD + d] + sc * ds);
  float y = old_v[row * kD + d] + sc * dv;
  float mean = blockSum256(y, sm) * (1.f / kD);
  float c = y - mean;
  float var = blockSum256(c * c, sm) * (1.f / kD);
  out_v[row * kD + d] = c * rsqrtf(var + kEps) * g[d] + bb[d];
}

// ---------------- small utilities ----------------------------------------
__global__ void k_conv_bf16(__bf16* __restrict__ dst, const float* __restrict__ src,
                            long n, const float* __restrict__ scaleVec)
{
  long i = (long)blockIdx.x * blockDim.x + threadIdx.x;
  const long stride = (long)gridDim.x * blockDim.x;
  for (; i < n; i += stride) {
    float v = src[i];
    if (scaleVec) v *= scaleVec[i & (kD - 1)];
    dst[i] = (__bf16)v;
  }
}

// W (256 x M, f32) -> bf16 W^T (M x 256); 32x32 LDS tile transpose
__global__ __launch_bounds__(256)
void k_transpose_bf16(__bf16* __restrict__ dst, const float* __restrict__ src, int M)
{
  __shared__ float t[32][33];
  const int m0 = blockIdx.x * 32, k0 = blockIdx.y * 32;
  const int tx = threadIdx.x & 31, ty = threadIdx.x >> 5;
#pragma unroll
  for (int j = 0; j < 4; ++j)
    t[ty + 8*j][tx] = src[(long)(k0 + ty + 8*j) * M + m0 + tx];
  __syncthreads();
#pragma unroll
  for (int j = 0; j < 4; ++j)
    dst[(long)(m0 + ty + 8*j) * kD + k0 + tx] = (__bf16)t[tx][ty + 8*j];
}

__global__ void k_tanh_inplace(float* __restrict__ x, long n)
{
  long i = (long)blockIdx.x * blockDim.x + threadIdx.x;
  const long stride = (long)gridDim.x * blockDim.x;
  for (; i < n; i += stride) x[i] = tanhf(x[i]);
}

__global__ __launch_bounds__(256)
void k_ln_inplace(float* __restrict__ x, const float* __restrict__ g,
                  const float* __restrict__ bb)
{
  __shared__ float sm[8];
  const long row = blockIdx.x;
  const int  d   = threadIdx.x;
  float y = x[row * kD + d];
  float mean = blockSum256(y, sm) * (1.f / kD);
  float c = y - mean;
  float var = blockSum256(c * c, sm) * (1.f / kD);
  x[row * kD + d] = c * rsqrtf(var + kEps) * g[d] + bb[d];
}

__global__ __launch_bounds__(256)
void k_bcast_row(float* __restrict__ dst, const float* __restrict__ src)
{
  dst[(long)blockIdx.x * kD + threadIdx.x] = src[threadIdx.x];
}

// ---------------- host orchestration --------------------------------------
extern "C" void kernel_launch(void* const* d_in, const int* in_sizes, int n_in,
                              void* d_out, int out_size, void* d_ws, size_t ws_size,
                              hipStream_t stream)
{
  (void)in_sizes; (void)n_in; (void)out_size; (void)ws_size;
  const float* s_state  = (const float*)d_in[0];
  const float* s_val    = (const float*)d_in[1];
  const float* w_pair_s = (const float*)d_in[2];
  const float* w_pair_e = (const float*)d_in[3];
  const float* w_pair_c = (const float*)d_in[4];
  const float* W_expand = (const float*)d_in[5];
  const float* b_expand = (const float*)d_in[6];
  const float* W_b2s    = (const float*)d_in[7];
  const float* b_b2s    = (const float*)d_in[8];
  const float* W_comp   = (const float*)d_in[9];
  const float* b_comp   = (const float*)d_in[10];
  const float* W_s2b    = (const float*)d_in[11];
  const float* b_s2b    = (const float*)d_in[12];
  const float* ln_g_s   = (const float*)d_in[13];
  const float* ln_b_s   = (const float*)d_in[14];
  const float* ln_g_e   = (const float*)d_in[15];
  const float* ln_b_e   = (const float*)d_in[16];
  const float* ln_g_c   = (const float*)d_in[17];
  const float* ln_b_c   = (const float*)d_in[18];

  const size_t sS = (size_t)kB * kS * kD;   // 8388608
  const size_t sE = (size_t)kB * kNE * kD;  // 2097152
  const size_t sC = (size_t)kB * kNC * kD;  // 524288

  char* p = (char*)d_ws;
  float* w_sst = (float*)p; p += sS * 4;    // s_state after window
  float* w_svl = (float*)p; p += sS * 4;    // s_val after window
  float* e_st1 = (float*)p; p += sE * 4;
  float* e_vl1 = (float*)p; p += sE * 4;
  float* e_st2 = (float*)p; p += sE * 4;
  float* e_vl2 = (float*)p; p += sE * 4;
  float* c_st0 = (float*)p; p += sC * 4;
  float* c_vl0 = (float*)p; p += sC * 4;
  float* acc_ns = (float*)p; p += sC * 4;
  float* acc_nv = (float*)p; p += sC * 4;
  __bf16* bfA = (__bf16*)p; p += sS * 2;
  __bf16* bfB = (__bf16*)p; p += (size_t)kD * kS * 2;  // 2.1M elems max
  int*   tidx = (int*)p;   p += (size_t)kB * kS * 4 * 4;
  float* tw   = (float*)p; p += (size_t)kB * kS * 4 * 4;

  float* o_ss = (float*)d_out;
  float* o_sv = o_ss + sS;
  float* o_ns = o_sv + sS;
  float* o_nv = o_ns + sC;

  // 1) window propagation + tanh/LN updates of sequence level
  k_window<<<kB * kS / 8, 256, 0, stream>>>(s_state, s_val, w_pair_s,
                                            ln_g_s, ln_b_s, w_sst, w_svl);

  // 2) expand transition (c -> e); c_state = 0, c_val = LN(0) = ln_b_c
  hipMemsetAsync(e_st1, 0, sE * 4, stream);
  hipMemsetAsync(e_vl1, 0, sE * 4, stream);
  hipMemsetAsync(c_st0, 0, sC * 4, stream);
  k_bcast_row<<<kB * kNC, 256, 0, stream>>>(c_vl0, ln_b_c);
  hipMemsetAsync(bfA, 0, sC * 2, stream);                       // bf16(c_state)=0
  k_transpose_bf16<<<dim3(kNE / 32, kD / 32), 256, 0, stream>>>(bfB, W_expand, kNE);
  k_gemm_topk<<<dim3(kNC / 64, kB), 128, 0, stream>>>(bfA, bfB, b_expand, 1.0f,
                                                      kNC, kNE, 1, tidx, tw);
  k_scatter<<<kB * kNC, 256, 0, stream>>>(c_st0, c_vl0, tidx, tw, kScB,
                                          kNC, kNE, e_st1, e_vl1);

  // 3) top-k propagation on expanded level
  k_conv_bf16<<<2048, 256, 0, stream>>>(bfA, e_st1, (long)sE, w_pair_e);
  k_conv_bf16<<<2048, 256, 0, stream>>>(bfB, e_st1, (long)sE, nullptr);
  k_gemm_topk<<<dim3(kNE / 64, kB), 128, 0, stream>>>(bfA, bfB, nullptr, kInvSqrtD,
                                                      kNE, kNE, 0, tidx, tw);
  k_gather_combine<<<kB * kNE, 256, 0, stream>>>(e_st1, e_vl1, e_st1, e_vl1,
                                                 tidx, tw, kScB, ln_g_e, ln_b_e,
                                                 kNE, e_st2, e_vl2);

  // 4) e -> s transition (biggest GEMM: 2048x8192x256 per batch)
  k_conv_bf16<<<2048, 256, 0, stream>>>(bfA, e_st2, (long)sE, nullptr);
  k_transpose_bf16<<<dim3(kS / 32, kD / 32), 256, 0, stream>>>(bfB, W_b2s, kS);
  k_gemm_topk<<<dim3(kNE / 64, kB), 128, 0, stream>>>(bfA, bfB, b_b2s, 1.0f,
                                                      kNE, kS, 1, tidx, tw);
  hipMemcpyAsync(o_ss, w_sst, sS * 4, hipMemcpyDeviceToDevice, stream);
  hipMemcpyAsync(o_sv, w_svl, sS * 4, hipMemcpyDeviceToDevice, stream);
  k_scatter<<<kB * kNE, 256, 0, stream>>>(e_st2, e_vl2, tidx, tw, kScX,
                                          kNE, kS, o_ss, o_sv);
  k_tanh_inplace<<<2048, 256, 0, stream>>>(o_ss, (long)sS);
  k_ln_inplace<<<kB * kS, 256, 0, stream>>>(o_sv, ln_g_s, ln_b_s);

  // 5) e -> c transition (bfA still holds bf16(e_st2))
  k_transpose_bf16<<<dim3(kNC / 32, kD / 32), 256, 0, stream>>>(bfB, W_comp, kNC);
  k_gemm_topk<<<dim3(kNE / 64, kB), 128, 0, stream>>>(bfA, bfB, b_comp, 1.0f,
                                                      kNE, kNC, 1, tidx, tw);
  hipMemsetAsync(acc_ns, 0, sC * 4, stream);
  k_bcast_row<<<kB * kNC, 256, 0, stream>>>(acc_nv, ln_b_c);
  k_scatter<<<kB * kNE, 256, 0, stream>>>(e_st2, e_vl2, tidx, tw, kScB,
                                          kNE, kNC, acc_ns, acc_nv);

  // 6) s -> c transition (uses updated s state/val)
  k_conv_bf16<<<2048, 256, 0, stream>>>(bfA, o_ss, (long)sS, nullptr);
  k_transpose_bf16<<<dim3(kNC / 32, kD / 32), 256, 0, stream>>>(bfB, W_s2b, kNC);
  k_gemm_topk<<<dim3(kS / 64, kB), 128, 0, stream>>>(bfA, bfB, b_s2b, 1.0f,
                                                     kS, kNC, 1, tidx, tw);
  k_scatter<<<kB * kS, 256, 0, stream>>>(o_ss, o_sv, tidx, tw, kScSB,
                                         kS, kNC, acc_ns, acc_nv);

  // 7) top-k propagation on compressed level -> final nc outputs
  k_conv_bf16<<<512, 256, 0, stream>>>(bfA, acc_ns, (long)sC, w_pair_c);
  k_conv_bf16<<<512, 256, 0, stream>>>(bfB, acc_ns, (long)sC, nullptr);
  k_gemm_topk<<<dim3(kNC / 64, kB), 128, 0, stream>>>(bfA, bfB, nullptr, kInvSqrtD,
                                                      kNC, kNC, 0, tidx, tw);
  k_gather_combine<<<kB * kNC, 256, 0, stream>>>(acc_ns, acc_nv, acc_ns, acc_nv,
                                                 tidx, tw, kScB, ln_g_c, ln_b_c,
                                                 kNC, o_ns, o_nv);
}
